// Evolving_GConv_36077725286756
// MI455X (gfx1250) — matrix operand order repaired and checked by reading, not compiled, not run on hardware
//
#include <hip/hip_runtime.h>
#include <hip/hip_bf16.h>
#include <math.h>

typedef __attribute__((ext_vector_type(16))) _Float16 v16h;
typedef __attribute__((ext_vector_type(8)))  _Float16 v8h;
typedef __attribute__((ext_vector_type(8)))  float    v8f;
typedef __attribute__((ext_vector_type(4)))  unsigned int u32x4;
typedef __attribute__((ext_vector_type(4)))  int      i32x4;
typedef __attribute__((ext_vector_type(8)))  int      i32x8;

union F16Frag { v16h v; struct { v8h lo, hi; } s; };

#define B_   16
#define C_   32
#define N_   300
#define T_   12
#define ST_  40
#define DH_  16
#define CO_  32
#define DI_  3      // DY_INTERVAL
#define NC_  4      // T / DI
#define ALPHA_ 0.05f

#define NPV  304    // N padded to multiple of 16 (M dim of adj GEMM)
#define NPW  320    // N padded to multiple of 32 (K dim of adj GEMM)
#define HCOL 96     // C * DI (N dim of adj GEMM / K dim of MLP GEMM)
#define HOROWS 912  // N*DI = 900 padded to 16

__device__ __forceinline__ float sigmoidf_(float x){ return 1.0f/(1.0f+__expf(-x)); }

// -------- states0 = st_node_fea @ W_s2d + b, broadcast over batch ----------
__global__ void k_states0(const float* __restrict__ stf, const float* __restrict__ W,
                          const float* __restrict__ bias, float* __restrict__ states){
  int i = blockIdx.x*blockDim.x + threadIdx.x;
  if (i >= N_*DH_) return;
  int n = i / DH_, d = i % DH_;
  float s = bias[d];
  for (int k = 0; k < ST_; k++) s += stf[n*ST_+k] * W[k*DH_+d];
  for (int b = 0; b < B_; b++) states[((size_t)b*N_+n)*DH_+d] = s;
}

// -------- W_mlp (96x32) -> transposed f16 Wt[o][c] (K-contiguous B frags) --
__global__ void k_convw(const float* __restrict__ W, _Float16* __restrict__ Wt){
  int i = blockIdx.x*blockDim.x + threadIdx.x;
  if (i >= HCOL*CO_) return;
  int o = i / HCOL, c = i % HCOL;
  Wt[(size_t)o*HCOL + c] = (_Float16)W[(size_t)c*CO_ + o];
}

// -------- GRU cell + per-node edge-score projections -----------------------
__global__ void k_gru(const float* __restrict__ x, int ci,
                      float* __restrict__ states,
                      const float* __restrict__ W_rz, const float* __restrict__ b_rz,
                      const float* __restrict__ W_h,  const float* __restrict__ b_h,
                      const float* __restrict__ W1,   const float* __restrict__ W2,
                      float* __restrict__ p1, float* __restrict__ q1,
                      float* __restrict__ p2, float* __restrict__ q2){
  int i = blockIdx.x*blockDim.x + threadIdx.x;
  if (i >= B_*N_) return;
  int b = i / N_, n = i % N_;
  int t0 = ci*DI_;
  float inp[C_];
  for (int c = 0; c < C_; c++){
    const float* xp = x + (((size_t)b*C_+c)*N_+n)*T_ + t0;
    inp[c] = (xp[0]+xp[1]+xp[2]) * (1.0f/3.0f);
  }
  float st[DH_];
  for (int d = 0; d < DH_; d++) st[d] = states[(size_t)i*DH_+d];
  float r[DH_], z[DH_];
  for (int j = 0; j < DH_; j++){
    float ar = b_rz[j], az = b_rz[DH_+j];
    for (int k = 0; k < C_; k++){
      ar += inp[k]*W_rz[k*(2*DH_)+j];
      az += inp[k]*W_rz[k*(2*DH_)+DH_+j];
    }
    for (int k = 0; k < DH_; k++){
      ar += st[k]*W_rz[(C_+k)*(2*DH_)+j];
      az += st[k]*W_rz[(C_+k)*(2*DH_)+DH_+j];
    }
    r[j] = sigmoidf_(ar); z[j] = sigmoidf_(az);
  }
  float hs[DH_];
  for (int j = 0; j < DH_; j++){
    float a = b_h[j];
    for (int k = 0; k < C_; k++)  a += inp[k]*W_h[k*DH_+j];
    for (int k = 0; k < DH_; k++) a += (r[k]*st[k])*W_h[(C_+k)*DH_+j];
    float hq = tanhf(a);
    float ns = z[j]*st[j] + (1.0f - z[j])*hq;
    states[(size_t)i*DH_+j] = ns;
    hs[j] = ns > 0.f ? ns : 0.f;
  }
  for (int d = 0; d < DH_; d++){
    float a1=0.f, c1=0.f, a2=0.f, c2=0.f;
    for (int k = 0; k < DH_; k++){
      a1 += hs[k]*W1[k*DH_+d];          // Wa = W_fc2[:DH]
      c1 += hs[k]*W1[(DH_+k)*DH_+d];    // Wb = W_fc2[DH:]
      a2 += hs[k]*W2[k*DH_+d];
      c2 += hs[k]*W2[(DH_+k)*DH_+d];
    }
    p1[(size_t)i*DH_+d]=a1; q1[(size_t)i*DH_+d]=c1;
    p2[(size_t)i*DH_+d]=a2; q2[(size_t)i*DH_+d]=c2;
  }
}

// -------- adjacency: adj = support * sigmoid(mask), padded f16 -------------
__global__ void k_adj(const float* __restrict__ p1, const float* __restrict__ q1,
                      const float* __restrict__ p2, const float* __restrict__ q2,
                      const float* __restrict__ bf2_1, const float* __restrict__ wf1_1,
                      const float* __restrict__ bf1_1,
                      const float* __restrict__ bf2_2, const float* __restrict__ wf1_2,
                      const float* __restrict__ bf1_2,
                      _Float16* __restrict__ adj){
  long i = (long)blockIdx.x*blockDim.x + threadIdx.x;
  const long tot = (long)B_*NPV*NPW;
  if (i >= tot) return;
  int w = (int)(i % NPW); long rr = i / NPW;
  int v = (int)(rr % NPV); int b = (int)(rr / NPV);
  float out = 0.f;
  if (v < N_ && w < N_){
    const float* pv1 = p1 + ((size_t)b*N_+v)*DH_;
    const float* qw1 = q1 + ((size_t)b*N_+w)*DH_;
    const float* pv2 = p2 + ((size_t)b*N_+v)*DH_;
    const float* qw2 = q2 + ((size_t)b*N_+w)*DH_;
    float s1 = bf1_1[0], s2 = bf1_2[0];
    for (int d = 0; d < DH_; d++){
      float h1 = pv1[d]+qw1[d]+bf2_1[d]; h1 = h1>0.f?h1:0.f; s1 += h1*wf1_1[d];
      float h2 = pv2[d]+qw2[d]+bf2_2[d]; h2 = h2>0.f?h2:0.f; s2 += h2*wf1_2[d];
    }
    out = s1 * sigmoidf_(s2);
  }
  adj[((size_t)b*NPV+v)*NPW + w] = (_Float16)out;
}

// -------- pack hop input TRANSPOSED: Ht[b][col<96][w<NPW] (K-contiguous) ---
__global__ void k_packHt(const float* __restrict__ src, int sT, int t0,
                         _Float16* __restrict__ Ht){
  int i = blockIdx.x*blockDim.x + threadIdx.x;
  if (i >= B_*HCOL*NPW) return;
  int w = i % NPW; int rr = i / NPW;
  int col = rr % HCOL; int b = rr / HCOL;
  float v = 0.f;
  if (w < N_){
    int c = col / DI_, t = col % DI_;
    v = src[(((size_t)b*C_+c)*N_+w)*sT + t0 + t];
  }
  Ht[((size_t)b*HCOL + col)*NPW + w] = (_Float16)v;
}

// -------- graph hop: hout = a*x_i + (1-a) * adj @ H  (TDM->LDS + WMMA) -----
// One 6-wave workgroup per (mtile, b): wave w handles column tile w.
// The 16x320 f16 adj row-panel is staged once in LDS via the Tensor Data
// Mover; A-fragments then come from LDS (ds_load_b128), B-fragments from
// global (global_load_b128).
__global__ void __launch_bounds__(192)
k_spmm(const _Float16* __restrict__ adj, const _Float16* __restrict__ Ht,
       const float* __restrict__ x, int ci, float* __restrict__ hout){
  __shared__ __align__(16) _Float16 lds_A[16*NPW];   // 10 KB
  int mtile = blockIdx.x;           // 0..18
  int b     = blockIdx.y;
  int tid   = threadIdx.x;
  int wv    = tid >> 5;             // 0..5 -> ntile
  int lane  = tid & 31;
  int lo = lane & 15, hi = lane >> 4;

  const _Float16* Apanel = adj + ((size_t)b*NPV + (size_t)mtile*16)*NPW;

#if __has_builtin(__builtin_amdgcn_tensor_load_to_lds)
  if (wv == 0){
    unsigned long long ga = (unsigned long long)Apanel;
    // D# group0: count=1, lds_addr=0 (our array is the sole LDS object),
    // global_addr at bits[120:64], type=2 at bits[127:126].
    u32x4 g0 = { 1u, 0u, (unsigned)(ga & 0xffffffffu),
                 (unsigned)(((ga >> 32) & 0x1ffffffu) | (2u << 30)) };
    // D# group1: data_size=2B, tensor_dim0=NPW, tensor_dim1=16,
    // tile_dim0=NPW, tile_dim1=16, tensor_dim0_stride=NPW.
    i32x8 g1 = { (int)(1u << 16),
                 (int)(((unsigned)NPW) << 16),
                 (int)(16u << 16),
                 (int)(((unsigned)NPW) << 16),
                 16, NPW, 0, 0 };
    i32x4 gz = {0,0,0,0};
#if defined(__clang_major__) && (__clang_major__ >= 23)
    i32x8 gz8 = {0,0,0,0,0,0,0,0};
    __builtin_amdgcn_tensor_load_to_lds(g0, g1, gz, gz, gz8, 0);
#else
    __builtin_amdgcn_tensor_load_to_lds(g0, g1, gz, gz, 0);
#endif
    __builtin_amdgcn_s_wait_tensorcnt(0);
  }
#else
  for (int idx = tid; idx < (16*NPW)/8; idx += 192)
    ((v8h*)lds_A)[idx] = ((const v8h*)Apanel)[idx];
#endif
  __syncthreads();

  // Escape the LDS pointer into the asm (which may have written through it)
  // so the TDM-written contents can't be folded to undef — but keep the
  // actual reads on the __shared__ object so they lower to ds_load_b128,
  // not flat_load.
  asm volatile("" : : "v"(lds_A) : "memory");

  const _Float16* Bt = Ht + ((size_t)b*HCOL + (size_t)(wv*16 + lo))*NPW;
  int kbA = hi*8, kbB = hi*8;
  v8f acc = {};
  for (int kt = 0; kt < NPW/32; kt++){
    int kb = kt*32;
    F16Frag a, bb;
    a.s.lo  = *(const v8h*)(lds_A + lo*NPW + kb + kbA);        // K = kbA..+7
    a.s.hi  = *(const v8h*)(lds_A + lo*NPW + kb + 16 + kbA);   // K = 16+kbA..
    bb.s.lo = *(const v8h*)(Bt + kb + kbB);                    // K = kbB..+7
    bb.s.hi = *(const v8h*)(Bt + kb + 16 + kbB);
    acc = __builtin_amdgcn_wmma_f32_16x16x32_f16(false, a.v, false, bb.v,
                                                 (short)0, acc, false, false);
  }
  int col = wv*16 + lo;
  int c = col / DI_, t = col % DI_;
  int t0 = ci*DI_;
  #pragma unroll
  for (int r = 0; r < 8; r++){
    int m = mtile*16 + r + hi*8;
    if (m < N_){
      float xi = x[(((size_t)b*C_+c)*N_+m)*T_ + t0 + t];
      hout[(((size_t)b*C_+c)*N_+m)*DI_ + t] = ALPHA_*xi + (1.0f-ALPHA_)*acc[r];
    }
  }
}

// -------- pack [x_i | h1 | h2] rows=(n,t) cols=c(96) to f16, row-major -----
__global__ void k_packHO(const float* __restrict__ x, int ci,
                         const float* __restrict__ h1, const float* __restrict__ h2,
                         _Float16* __restrict__ HO){
  int i = blockIdx.x*blockDim.x + threadIdx.x;
  if (i >= B_*HOROWS*HCOL) return;
  int c = i % HCOL; int rr = i / HCOL;
  int row = rr % HOROWS; int b = rr / HOROWS;
  float v = 0.f;
  if (row < N_*DI_){
    int n = row / DI_, t = row % DI_;
    int seg = c / C_, cc = c % C_;
    if (seg == 0) v = x[(((size_t)b*C_+cc)*N_+n)*T_ + ci*DI_ + t];
    else {
      const float* h = (seg == 1) ? h1 : h2;
      v = h[(((size_t)b*C_+cc)*N_+n)*DI_ + t];
    }
  }
  HO[((size_t)b*HOROWS + row)*HCOL + c] = (_Float16)v;
}

// -------- output MLP: out = HO @ W_mlp + b  (WMMA, all-b128 frags) ---------
__global__ void k_mlp(const _Float16* __restrict__ HO, const _Float16* __restrict__ Wt,
                      const float* __restrict__ b_mlp, int ci, float* __restrict__ out){
  int ntile = blockIdx.x;         // 0..1  (32/16 cols)
  int mtile = blockIdx.y;         // 0..56 (912/16 rows)
  int b     = blockIdx.z;
  int lane  = threadIdx.x;
  int lo = lane & 15, hi = lane >> 4;
  v8f acc = {};
  const _Float16* Ab = HO + ((size_t)b*HOROWS + (size_t)mtile*16 + lo)*HCOL;
  const _Float16* Bb = Wt + (size_t)(ntile*16 + lo)*HCOL;
  int kbA = hi*8, kbB = hi*8;
  for (int kt = 0; kt < HCOL/32; kt++){
    int kb = kt*32;
    F16Frag a, bb;
    a.s.lo  = *(const v8h*)(Ab + kb + kbA);
    a.s.hi  = *(const v8h*)(Ab + kb + 16 + kbA);
    bb.s.lo = *(const v8h*)(Bb + kb + kbB);
    bb.s.hi = *(const v8h*)(Bb + kb + 16 + kbB);
    acc = __builtin_amdgcn_wmma_f32_16x16x32_f16(false, a.v, false, bb.v,
                                                 (short)0, acc, false, false);
  }
  int col = ntile*16 + lo;
  float bias = b_mlp[col];
  #pragma unroll
  for (int r = 0; r < 8; r++){
    int m = mtile*16 + r + hi*8;
    if (m < N_*DI_){
      int n = m / DI_, t = m % DI_;
      out[(((size_t)b*CO_+col)*N_+n)*T_ + ci*DI_ + t] = acc[r] + bias;
    }
  }
}

extern "C" void kernel_launch(void* const* d_in, const int* in_sizes, int n_in,
                              void* d_out, int out_size, void* d_ws, size_t ws_size,
                              hipStream_t stream){
  (void)in_sizes; (void)n_in; (void)out_size; (void)ws_size;
  const float* x     = (const float*)d_in[0];
  const float* stf   = (const float*)d_in[1];
  const float* W_s2d = (const float*)d_in[2];
  const float* b_s2d = (const float*)d_in[3];
  const float* W_rz  = (const float*)d_in[4];
  const float* b_rz  = (const float*)d_in[5];
  const float* W_h   = (const float*)d_in[6];
  const float* b_h   = (const float*)d_in[7];
  const float* W1f2  = (const float*)d_in[8];
  const float* b1f2  = (const float*)d_in[9];
  const float* W1f1  = (const float*)d_in[10];
  const float* b1f1  = (const float*)d_in[11];
  const float* W2f2  = (const float*)d_in[12];
  const float* b2f2  = (const float*)d_in[13];
  const float* W2f1  = (const float*)d_in[14];
  const float* b2f1  = (const float*)d_in[15];
  const float* W_mlp = (const float*)d_in[16];
  const float* b_mlp = (const float*)d_in[17];
  float* out = (float*)d_out;

  char* ws = (char*)d_ws;
  size_t off = 0;
  auto alloc = [&](size_t bytes)->char*{
    char* p = ws + off; off += (bytes + 255) & ~(size_t)255; return p;
  };
  float*     states = (float*)    alloc((size_t)B_*N_*DH_*4);
  float*     p1     = (float*)    alloc((size_t)B_*N_*DH_*4);
  float*     q1     = (float*)    alloc((size_t)B_*N_*DH_*4);
  float*     p2     = (float*)    alloc((size_t)B_*N_*DH_*4);
  float*     q2     = (float*)    alloc((size_t)B_*N_*DH_*4);
  _Float16*  adj    = (_Float16*) alloc((size_t)B_*NPV*NPW*2);
  _Float16*  Ht     = (_Float16*) alloc((size_t)B_*HCOL*NPW*2);
  float*     h1     = (float*)    alloc((size_t)B_*C_*N_*DI_*4);
  float*     h2     = (float*)    alloc((size_t)B_*C_*N_*DI_*4);
  _Float16*  HO     = (_Float16*) alloc((size_t)B_*HOROWS*HCOL*2);
  _Float16*  Wt     = (_Float16*) alloc((size_t)HCOL*CO_*2);

  k_states0<<<(N_*DH_+255)/256, 256, 0, stream>>>(stf, W_s2d, b_s2d, states);
  k_convw<<<(HCOL*CO_+255)/256, 256, 0, stream>>>(W_mlp, Wt);

  const long adjtot = (long)B_*NPV*NPW;
  for (int ci = 0; ci < NC_; ci++){
    k_gru<<<(B_*N_+127)/128, 128, 0, stream>>>(x, ci, states, W_rz, b_rz, W_h, b_h,
                                               W1f2, W2f2, p1, q1, p2, q2);
    k_adj<<<(unsigned)((adjtot+255)/256), 256, 0, stream>>>(p1, q1, p2, q2,
                                               b1f2, W1f1, b1f1, b2f2, W2f1, b2f1, adj);
    // hop 1: Ht <- x_i ; h1 <- a*x_i + (1-a)*adj@H
    k_packHt<<<(B_*HCOL*NPW+255)/256, 256, 0, stream>>>(x, T_, ci*DI_, Ht);
    k_spmm<<<dim3(NPV/16, B_), 192, 0, stream>>>(adj, Ht, x, ci, h1);
    // hop 2: Ht <- h1 ; h2 <- a*x_i + (1-a)*adj@H
    k_packHt<<<(B_*HCOL*NPW+255)/256, 256, 0, stream>>>(h1, DI_, 0, Ht);
    k_spmm<<<dim3(NPV/16, B_), 192, 0, stream>>>(adj, Ht, x, ci, h2);
    // output MLP over concat([x_i, h1, h2])
    k_packHO<<<(B_*HOROWS*HCOL+255)/256, 256, 0, stream>>>(x, ci, h1, h2, HO);
    k_mlp<<<dim3(CO_/16, HOROWS/16, B_), 32, 0, stream>>>(HO, Wt, b_mlp, ci, out);
  }
}
